// Module_24008867185192
// MI455X (gfx1250) — matrix operand order repaired and checked by reading, not compiled, not run on hardware
//
#include <hip/hip_runtime.h>
#include <hip/hip_bf16.h>

typedef __attribute__((ext_vector_type(16))) _Float16 v16h;
typedef __attribute__((ext_vector_type(8)))  _Float16 h8;
typedef __attribute__((ext_vector_type(4)))  _Float16 h4;
typedef __attribute__((ext_vector_type(8)))  float    v8f;
typedef int gv4i __attribute__((vector_size(16)));   // matches builtin param type

#define ROWLEN  20001          // interactions row length (items+1)
#define KBIG    20000          // reduction length of the two big GEMMs
#define BATCH   4096
#define KSTEP   32

// ---- CDNA5 async global->LDS copies (guarded; sync fallback keeps compile safe)
#if defined(__AMDGCN__) && defined(__has_builtin)
#if __has_builtin(__builtin_amdgcn_global_load_async_to_lds_b128) && \
    __has_builtin(__builtin_amdgcn_global_load_async_to_lds_b32)
#define HAVE_ASYNC_LDS 1
#endif
#endif
#ifndef HAVE_ASYNC_LDS
#define HAVE_ASYNC_LDS 0
#endif

#define AS_GLOBAL __attribute__((address_space(1)))
#define AS_LDS    __attribute__((address_space(3)))

#if HAVE_ASYNC_LDS
__device__ __forceinline__ void async_ld_b128(const void* gp, void* lp) {
  __builtin_amdgcn_global_load_async_to_lds_b128(
      (AS_GLOBAL gv4i*)gp, (AS_LDS gv4i*)lp, 0, 0);
}
__device__ __forceinline__ void async_ld_b32(const void* gp, void* lp) {
  __builtin_amdgcn_global_load_async_to_lds_b32(
      (AS_GLOBAL int*)gp, (AS_LDS int*)lp, 0, 0);
}
#endif

__device__ __forceinline__ void wait_async_all() {
#if HAVE_ASYNC_LDS
#if __has_builtin(__builtin_amdgcn_s_wait_asynccnt)
  __builtin_amdgcn_s_wait_asynccnt(0);
#else
  asm volatile("s_wait_asynccnt 0x0" ::: "memory");
#endif
#endif
}

// ---------------------------------------------------------------------------
// Elementwise f32 -> f16 convert (weights stay in native torch (N x K) layout,
// so GEMM LDS staging is a straight contiguous copy -- no transpose anywhere).
// ---------------------------------------------------------------------------
__global__ __launch_bounds__(256) void k_convert_f16(
    const float* __restrict__ W, _Float16* __restrict__ Wh, size_t total4) {
  size_t idx = (size_t)blockIdx.x * 256 + threadIdx.x;
  if (idx >= total4) return;
  const float4 v = ((const float4*)W)[idx];
  h4 o = {(_Float16)v.x, (_Float16)v.y, (_Float16)v.z, (_Float16)v.w};
  ((h4*)Wh)[idx] = o;
}

// ---------------------------------------------------------------------------
// Fused binary-A GEMM:  C(4096 x 256) = A(4096 x 20000, {0,1}) x W^T
// W kept native (256 x 20000) f16.
// MODE 0 (user path): A[b,k] = inter[uidx[b]*ROWLEN + k], zeroed at k==iidx[b];
//                     NT interaction loads; writes f16 output (feeds MLP).
// MODE 1 (item path): A[b,k] = inter[k*ROWLEN + iidx[b]], zeroed at k==uidx[b];
//                     RT loads (rows shared across blocks via L2); f32 output.
// 256 threads = 8 waves; tile 32(M) x 256(N); K stepped by 32.
// Double-buffered LDS: stage(s+1) via ASYNC-to-LDS overlaps WMMA of step s;
// one barrier per K-step.
// ---------------------------------------------------------------------------
template<int MODE>
__global__ __launch_bounds__(256) void k_spbin_gemm(
    const unsigned char* __restrict__ inter,
    const int* __restrict__ uidx,
    const int* __restrict__ iidx,
    const _Float16* __restrict__ Wp,     // (256 x 20000) f16, native layout
    float* __restrict__ Cf32,            // MODE 1 output
    _Float16* __restrict__ Cf16)         // MODE 0 output
{
  constexpr int MT    = 32;
  constexpr int NSTEP = KBIG / KSTEP;    // 625
  __shared__ _Float16 As[2][MT * 40];    // [m][k], stride 40 halves (pad)
  __shared__ _Float16 Bs[2][256 * 40];   // [n][k], stride 40 halves

  const int tid  = threadIdx.x;
  const int lane = tid & 31;
  const int wave = tid >> 5;
  const int g    = lane >> 4;            // lane half-group
  const int ln   = lane & 15;
  const int wm   = wave & 1;             // 2 M-blocks of 16 rows
  const int wn   = wave >> 1;            // 4 N-blocks of 64 cols
  const int m0   = blockIdx.x * MT;

  const int arow = tid >> 3;             // MODE0: A row 0..31 | MODE1: k 0..31
  const int asub = (tid & 7) * 4;        // MODE0: k chunk     | MODE1: m chunk

  size_t abase = 0; int zcol = -1;
  int icol[4] = {0, 0, 0, 0}; int urow[4] = {-1, -1, -1, -1};
  if (MODE == 0) {
    abase = (size_t)uidx[m0 + arow] * ROWLEN;
    zcol  = iidx[m0 + arow];
  } else {
#pragma unroll
    for (int j = 0; j < 4; ++j) {
      icol[j] = iidx[m0 + asub + j];
      urow[j] = uidx[m0 + asub + j];
    }
  }

  // stage one K-step into LDS buffer `buf`
  auto stage = [&](int k0, int buf) {
    {
      _Float16* Bsb = &Bs[buf][0];
      const int nsub = tid >> 2;          // 0..63
      const int c    = (tid & 3) * 8;     // 0,8,16,24 (halves)
#pragma unroll
      for (int np = 0; np < 256; np += 64) {
        const int n = np + nsub;
        const _Float16* gp = Wp + (size_t)n * KBIG + k0 + c;
#if HAVE_ASYNC_LDS
        async_ld_b128(gp, &Bsb[n * 40 + c]);
#else
        *(h8*)&Bsb[n * 40 + c] = *(const h8*)gp;
#endif
      }
    }
    _Float16* Asb = &As[buf][0];
    if (MODE == 0) {
      const unsigned char* p = inter + abase + k0 + asub;
      __builtin_prefetch(p + KSTEP, 0, 1);     // next K chunk
#pragma unroll
      for (int j = 0; j < 4; ++j) {
        const int kk = k0 + asub + j;
        const unsigned char bv = __builtin_nontemporal_load(p + j);
        Asb[arow * 40 + asub + j] =
            (bv && kk != zcol) ? (_Float16)1.0f : (_Float16)0.0f;
      }
    } else {
      const int kk = k0 + arow;
      const unsigned char* prow = inter + (size_t)kk * ROWLEN;
#pragma unroll
      for (int j = 0; j < 4; ++j) {
        Asb[(asub + j) * 40 + arow] =
            (prow[icol[j]] && kk != urow[j]) ? (_Float16)1.0f : (_Float16)0.0f;
      }
    }
  };

  v8f acc[4] = {};

  stage(0, 0);                           // prologue
  for (int s = 0; s < NSTEP; ++s) {
    wait_async_all();                    // step-s asyncs (issued last iter) done
    __syncthreads();                     // buf[s&1] published to all waves
    if (s + 1 < NSTEP) stage((s + 1) * KSTEP, (s + 1) & 1);  // overlaps compute

    const _Float16* Asb = &As[s & 1][0];
    const _Float16* Bsb = &Bs[s & 1][0];

    // ---- WMMA: A frag = two b128 LDS loads; each B frag = two b128 loads ----
    v16h af;
    {
      const _Float16* ap = Asb + (wm * 16 + ln) * 40 + g * 8;
      ((h8*)&af)[0] = *(const h8*)(ap);        // k = g*8 + 0..7
      ((h8*)&af)[1] = *(const h8*)(ap + 16);   // k = 16 + g*8 + 0..7
    }
#pragma unroll
    for (int t = 0; t < 4; ++t) {
      const int n = wn * 64 + t * 16 + ln;
      const _Float16* bp = Bsb + n * 40 + g * 8;
      v16h bf;
      ((h8*)&bf)[0] = *(const h8*)(bp);        // k = g*8 + 0..7
      ((h8*)&bf)[1] = *(const h8*)(bp + 16);   // k = 16 + g*8 + 0..7
      acc[t] = __builtin_amdgcn_wmma_f32_16x16x32_f16(
          false, af, false, bf, (short)0, acc[t], false, false);
    }
  }

  // ---- store D (VGPR r -> M = r + 8*g) ----
#pragma unroll
  for (int t = 0; t < 4; ++t) {
    const int col = wn * 64 + t * 16 + ln;
#pragma unroll
    for (int r = 0; r < 8; ++r) {
      const int row = m0 + wm * 16 + g * 8 + r;
      if (MODE == 0) Cf16[(size_t)row * 256 + col] = (_Float16)acc[t][r];
      else           Cf32[(size_t)row * 256 + col] = acc[t][r];
    }
  }
}

// ---------------------------------------------------------------------------
// Fused  Y = ReLU(LN(X @ W^T + bias))  via WMMA; W native (NDIM x KDIM) f16.
// Workgroup: 16 rows x NDIM (full row width so LN reduces locally).
// LDS union: GEMM tiles (phase 1) reuse the same space as the f32 staging /
// reduction buffers (phase 2) -> <= 42 KB LDS.
// ---------------------------------------------------------------------------
template<int KDIM, int NDIM, bool OUT16>
__global__ __launch_bounds__(256) void k_mlp_ln_relu(
    const _Float16* __restrict__ X,      // (4096 x KDIM) f16
    const _Float16* __restrict__ Wp,     // (NDIM x KDIM) f16
    const float* __restrict__ bias,
    const float* __restrict__ gamma,
    const float* __restrict__ beta,
    _Float16* __restrict__ Y16,
    float* __restrict__ Y32)
{
  constexpr int NW = NDIM / 8;           // cols per wave
  constexpr int T  = NW / 16;            // wmma tiles per wave
  __shared__ union SU {
    struct { _Float16 As[16 * 40]; _Float16 Bs[NDIM * 40]; } p1;
    struct { float Out[16 * NDIM]; float RS[16][16]; float RQ[16][16];
             float Mean[16]; float Istd[16]; } p2;
  } sm;

  const int tid  = threadIdx.x;
  const int lane = tid & 31, wave = tid >> 5;
  const int g    = lane >> 4, ln = lane & 15;
  const int m0   = blockIdx.x * 16;

  v8f acc[T] = {};

  for (int k0 = 0; k0 < KDIM; k0 += KSTEP) {
    {   // A: 16 x 32 halves, one dword per thread (async)
      const int m = tid >> 4;
      const int k = (tid & 15) * 2;
      const _Float16* gp = X + (size_t)(m0 + m) * KDIM + k0 + k;
      _Float16* lp = &sm.p1.As[m * 40 + k];
#if HAVE_ASYNC_LDS
      async_ld_b32(gp, lp);
#else
      *(unsigned int*)lp = *(const unsigned int*)gp;
#endif
    }
    {   // B: NDIM x 32 halves, b128 per lane (async)
      const int nsub = tid >> 2;
      const int c    = (tid & 3) * 8;
#pragma unroll
      for (int np = 0; np < NDIM; np += 64) {
        const int n = np + nsub;
        const _Float16* gp = Wp + (size_t)n * KDIM + k0 + c;
        _Float16* lp = &sm.p1.Bs[n * 40 + c];
#if HAVE_ASYNC_LDS
        async_ld_b128(gp, lp);
#else
        *(h8*)lp = *(const h8*)gp;
#endif
      }
    }
    wait_async_all();
    __syncthreads();
    v16h af;
    {
      const _Float16* ap = sm.p1.As + ln * 40 + g * 8;
      ((h8*)&af)[0] = *(const h8*)(ap);
      ((h8*)&af)[1] = *(const h8*)(ap + 16);
    }
#pragma unroll
    for (int t = 0; t < T; ++t) {
      const int n = wave * NW + t * 16 + ln;
      const _Float16* bp = sm.p1.Bs + n * 40 + g * 8;
      v16h bf;
      ((h8*)&bf)[0] = *(const h8*)(bp);
      ((h8*)&bf)[1] = *(const h8*)(bp + 16);
      acc[t] = __builtin_amdgcn_wmma_f32_16x16x32_f16(
          false, af, false, bf, (short)0, acc[t], false, false);
    }
    __syncthreads();          // also fences before LDS region is repurposed
  }

  // ---- stage f32 (+bias) to LDS ----
#pragma unroll
  for (int t = 0; t < T; ++t) {
    const int col = wave * NW + t * 16 + ln;
    const float bv = bias[col];
#pragma unroll
    for (int r = 0; r < 8; ++r)
      sm.p2.Out[(g * 8 + r) * NDIM + col] = acc[t][r] + bv;
  }
  __syncthreads();

  // ---- LayerNorm stats: 16 threads per row ----
  {
    const int row = tid >> 4, sub = tid & 15;
    float s = 0.f, q = 0.f;
    for (int n = sub; n < NDIM; n += 16) {
      const float x = sm.p2.Out[row * NDIM + n];
      s += x; q += x * x;
    }
    sm.p2.RS[row][sub] = s;
    sm.p2.RQ[row][sub] = q;
  }
  __syncthreads();
  if ((tid & 15) == 0) {
    const int row = tid >> 4;
    float s = 0.f, q = 0.f;
#pragma unroll
    for (int i = 0; i < 16; ++i) { s += sm.p2.RS[row][i]; q += sm.p2.RQ[row][i]; }
    const float mean = s / NDIM;
    float var = q / NDIM - mean * mean;
    var = var < 0.f ? 0.f : var;
    sm.p2.Mean[row] = mean;
    sm.p2.Istd[row] = rsqrtf(var + 1e-5f);
  }
  __syncthreads();

  // ---- normalize + affine + ReLU, write out ----
  for (int idx = tid; idx < 16 * NDIM; idx += 256) {
    const int row = idx / NDIM, n = idx - row * NDIM;
    float y = (sm.p2.Out[idx] - sm.p2.Mean[row]) * sm.p2.Istd[row] * gamma[n] + beta[n];
    y = y > 0.f ? y : 0.f;
    if (OUT16) Y16[(size_t)(m0 + row) * NDIM + n] = (_Float16)y;
    else       Y32[(size_t)(m0 + row) * NDIM + n] = y;
  }
}

// ---------------------------------------------------------------------------
// logit[b] = sum_h user_rep[b,h] * item_rep[b,h] * Wl[h] + bl
// ---------------------------------------------------------------------------
__global__ __launch_bounds__(256) void k_final(
    const float* __restrict__ ur, const float* __restrict__ ir,
    const float* __restrict__ Wl, const float* __restrict__ bl,
    float* __restrict__ out)
{
  __shared__ float red[256];
  const int b = blockIdx.x, t = threadIdx.x;
  red[t] = ur[(size_t)b * 256 + t] * ir[(size_t)b * 256 + t] * Wl[t];
  __syncthreads();
  for (int s = 128; s > 0; s >>= 1) {
    if (t < s) red[t] += red[t + s];
    __syncthreads();
  }
  if (t == 0) out[b] = red[0] + bl[0];
}

// ---------------------------------------------------------------------------
extern "C" void kernel_launch(void* const* d_in, const int* in_sizes, int n_in,
                              void* d_out, int out_size, void* d_ws, size_t ws_size,
                              hipStream_t stream) {
  const int*            uidx  = (const int*)d_in[0];
  const int*            iidx  = (const int*)d_in[1];
  const unsigned char*  inter = (const unsigned char*)d_in[2];   // bool matrix
  const float* Wu  = (const float*)d_in[3];
  const float* Wi  = (const float*)d_in[4];
  const float* W1  = (const float*)d_in[5];
  const float* b1  = (const float*)d_in[6];
  const float* g1  = (const float*)d_in[7];
  const float* be1 = (const float*)d_in[8];
  const float* W2  = (const float*)d_in[9];
  const float* b2  = (const float*)d_in[10];
  const float* g2  = (const float*)d_in[11];
  const float* be2 = (const float*)d_in[12];
  const float* Wl  = (const float*)d_in[13];
  const float* bl  = (const float*)d_in[14];
  float* out = (float*)d_out;

  // workspace carve-out (~35 MB total)
  char* ws = (char*)d_ws;
  size_t off = 0;
  auto alloc = [&](size_t bytes) {
    char* p = ws + off;
    off += (bytes + 255) & ~(size_t)255;
    return p;
  };
  _Float16* Wup  = (_Float16*)alloc((size_t)256 * KBIG * 2);
  _Float16* Wip  = (_Float16*)alloc((size_t)256 * KBIG * 2);
  _Float16* W1p  = (_Float16*)alloc((size_t)512 * 256 * 2);
  _Float16* W2p  = (_Float16*)alloc((size_t)256 * 512 * 2);
  _Float16* Uf16 = (_Float16*)alloc((size_t)BATCH * 256 * 2);
  float*    Irep = (float*)   alloc((size_t)BATCH * 256 * 4);
  _Float16* H1f  = (_Float16*)alloc((size_t)BATCH * 512 * 2);
  float*    Urep = (float*)   alloc((size_t)BATCH * 256 * 4);

  // weight prep: f16 convert, native (N x K) layout (no transpose needed)
  {
    const size_t t4_big = (size_t)256 * KBIG / 4;
    const size_t t4_mlp = (size_t)512 * 256 / 4;
    k_convert_f16<<<(unsigned)((t4_big + 255) / 256), 256, 0, stream>>>(Wu, Wup, t4_big);
    k_convert_f16<<<(unsigned)((t4_big + 255) / 256), 256, 0, stream>>>(Wi, Wip, t4_big);
    k_convert_f16<<<(unsigned)((t4_mlp + 255) / 256), 256, 0, stream>>>(W1, W1p, t4_mlp);
    k_convert_f16<<<(unsigned)((t4_mlp + 255) / 256), 256, 0, stream>>>(W2, W2p, t4_mlp);
  }

  // fused binary-A GEMMs (user row-gather, item column-gather)
  k_spbin_gemm<0><<<BATCH / 32, 256, 0, stream>>>(inter, uidx, iidx, Wup, nullptr, Uf16);
  k_spbin_gemm<1><<<BATCH / 32, 256, 0, stream>>>(inter, uidx, iidx, Wip, Irep, nullptr);

  // MLP: 256 -> 512 -> 256, each fused with LN + ReLU
  k_mlp_ln_relu<256, 512, true ><<<BATCH / 16, 256, 0, stream>>>(Uf16, W1p, b1, g1, be1, H1f, nullptr);
  k_mlp_ln_relu<512, 256, false><<<BATCH / 16, 256, 0, stream>>>(H1f, W2p, b2, g2, be2, nullptr, Urep);

  // pred = user_rep * item_rep; logit = pred @ Wl.T + bl
  k_final<<<BATCH, 256, 0, stream>>>(Urep, Irep, Wl, bl, out);
}
